// LovaszSoftmaxLoss_14216341750123
// MI455X (gfx1250) — compile-verified, matrix operand order checked by reading, not run on hardware
//
#include <hip/hip_runtime.h>
#include <hip/hip_bf16.h>
#include <stdint.h>

// ---------------------------------------------------------------------------
// Lovasz-Softmax loss for logits [8,19,512,1024] f32, labels [8,512,1024] i64.
// Sort-dominated, memory-bound workload. Strategy:
//   prep (softmax stats) -> per class: build sort keys -> 8x 4-bit stable LSD
//   radix sort (descending error) -> WMMA-based tile count reductions ->
//   scans -> jaccard-gradient dot product -> present-class average.
// WMMA f32_16x16x32_f16 with a ones B-matrix performs 512-element reductions
// of the {0,1} fs / vs-fs counts (exact in f16/f32).
// Workspace requirement: ~105 MB (layout below).
// ---------------------------------------------------------------------------

typedef __attribute__((ext_vector_type(16))) _Float16 v16h;
typedef __attribute__((ext_vector_type(8)))  float    v8f;

typedef unsigned long long u64;
typedef unsigned int       u32;

#define NUM_CLASSES 19
#define IGNORE_IDX  255u
#define HW_         524288      // 512*1024 = 2^19
#define HW_SHIFT    19
#define NPIX        4194304     // 8 * HW_
#define NTHREADS    256
#define TILE        4096        // 256 threads * 16 items
#define NB          (NPIX / TILE)   // 1024 tiles
#define RADIX_BITS  4
#define RADIX       16
#define N_PASSES    8           // sort key bits [33:2], 8 x 4-bit digits

// ---------------- workspace layout (bytes) ----------------
#define OFF_ITEMS0   0ull                      // u64[NPIX]  33,554,432
#define OFF_ITEMS1   33554432ull               // u64[NPIX]  33,554,432
#define OFF_GOFF     67108864ull               // u32[16*NB]     65,536
#define OFF_MMAX     67174400ull               // f32[NPIX]  16,777,216
#define OFF_MSUM     83951616ull               // f32[NPIX]  16,777,216
#define OFF_LAB8     100728832ull              // u8 [NPIX]   4,194,304
#define OFF_PARTF    104923136ull              // f32[NB]         4,096
#define OFF_PARTVF   104927232ull              // f32[NB]         4,096
#define OFF_LOSSBLK  104931328ull              // f32[NB]         4,096
#define OFF_GTS      104935424ull              // f32[19]
#define OFF_LOSSACC  104935680ull              // f32[19]
// total ~104,935,756 bytes

__device__ __forceinline__ unsigned digit_of(u64 item, int shift) {
  return (unsigned)((item >> shift) & (u64)(RADIX - 1));
}

// ---------------------------------------------------------------------------
// Per-pixel softmax statistics (max, sum of exp) + compressed labels.
// ---------------------------------------------------------------------------
__global__ void k_prep(const float* __restrict__ logits,
                       const long long* __restrict__ labels,
                       float* __restrict__ mmax, float* __restrict__ msum,
                       unsigned char* __restrict__ lab8) {
  int n  = blockIdx.x * NTHREADS + threadIdx.x;
  int b  = n >> HW_SHIFT;
  int hw = n & (HW_ - 1);
  const float* lp = logits + (size_t)b * NUM_CLASSES * HW_ + hw;
  float m = -3.4e38f;
#pragma unroll
  for (int c = 0; c < NUM_CLASSES; ++c) m = fmaxf(m, lp[(size_t)c * HW_]);
  float s = 0.f;
#pragma unroll
  for (int c = 0; c < NUM_CLASSES; ++c) s += __expf(lp[(size_t)c * HW_] - m);
  mmax[n] = m;
  msum[n] = s;
  lab8[n] = (unsigned char)labels[n];
}

// ---------------------------------------------------------------------------
// Build sort items for class c.
// item = (~bits(err))<<2 | fg<<1 | valid; ascending radix on bits[33:2]
// == descending error (err >= 0 so IEEE bits are order-preserving).
// ---------------------------------------------------------------------------
__global__ void k_build(const float* __restrict__ logits,
                        const float* __restrict__ mmax,
                        const float* __restrict__ msum,
                        const unsigned char* __restrict__ lab8,
                        u64* __restrict__ items, int c) {
  int n  = blockIdx.x * NTHREADS + threadIdx.x;
  int b  = n >> HW_SHIFT;
  int hw = n & (HW_ - 1);
  float l = logits[((size_t)b * NUM_CLASSES + (size_t)c) * HW_ + hw];
  float p = __expf(l - mmax[n]) / msum[n];
  unsigned lab   = lab8[n];
  unsigned valid = (lab != IGNORE_IDX) ? 1u : 0u;
  unsigned fg    = (lab == (unsigned)c) ? 1u : 0u;
  float err = valid ? fabsf((float)fg - p) : 0.f;
  u32 key = ~__float_as_uint(err);
  items[n] = ((u64)key << 2) | (u64)(fg << 1) | (u64)valid;
}

// ---------------------------------------------------------------------------
// Radix pass: per-tile digit histogram, column-major [digit][block].
// ---------------------------------------------------------------------------
__global__ void k_hist(const u64* __restrict__ in, u32* __restrict__ colHist,
                       int shift) {
  __shared__ u32 cnt[RADIX];
  int tid = threadIdx.x;
  if (tid < RADIX) cnt[tid] = 0u;
  __syncthreads();
  size_t base = (size_t)blockIdx.x * TILE;
  for (int r = 0; r < 16; ++r) {
    unsigned d = digit_of(in[base + (size_t)r * NTHREADS + tid], shift);
    atomicAdd(&cnt[d], 1u);
  }
  __syncthreads();
  if (tid < RADIX) colHist[(size_t)tid * NB + blockIdx.x] = cnt[tid];
}

// ---------------------------------------------------------------------------
// Single-block exclusive scan of u32 array (digit-major global offsets).
// ---------------------------------------------------------------------------
__global__ void k_scan_u32(u32* __restrict__ a, int M) {
  __shared__ u32 s[NTHREADS];
  __shared__ u32 carry;
  int tid = threadIdx.x;
  if (tid == 0) carry = 0u;
  __syncthreads();
  for (int off = 0; off < M; off += NTHREADS) {
    u32 v = a[off + tid];
    s[tid] = v;
    __syncthreads();
    for (int d = 1; d < NTHREADS; d <<= 1) {
      u32 t = (tid >= d) ? s[tid - d] : 0u;
      __syncthreads();
      s[tid] += t;
      __syncthreads();
    }
    u32 incl = s[tid];
    a[off + tid] = incl - v + carry;
    __syncthreads();
    if (tid == NTHREADS - 1) carry += incl;
    __syncthreads();
  }
}

// ---------------------------------------------------------------------------
// Radix pass: stable scatter. In-block ranks via wave32 ballot digit-match +
// per-(digit,round,wave) LDS counts + block exclusive scan. Tile index order
// is (round, wave, lane) which the scan order [d][r][w] preserves -> stable.
// ---------------------------------------------------------------------------
__global__ void k_scatter(const u64* __restrict__ in, u64* __restrict__ outp,
                          const u32* __restrict__ gOff, int shift) {
  __shared__ u32 wdc[RADIX * 16 * 8];   // [d][r][w] -> d*128 + r*8 + w
  __shared__ u32 dstart[RADIX];
  __shared__ u64 st[TILE];
  __shared__ u32 ssc[NTHREADS];
  __shared__ u32 scarry;
  int tid  = threadIdx.x;
  int lane = tid & 31;
  int w    = tid >> 5;
  for (int i = tid; i < RADIX * 128; i += NTHREADS) wdc[i] = 0u;
  if (tid == 0) scarry = 0u;
  __syncthreads();

  size_t base = (size_t)blockIdx.x * TILE;
  u64 it[16]; unsigned dg[16]; unsigned lr[16];
  for (int r = 0; r < 16; ++r) {
    u64 v = in[base + (size_t)r * NTHREADS + tid];
    it[r] = v;
    unsigned d = digit_of(v, shift);
    dg[r] = d;
    unsigned mask = 0xFFFFFFFFu;
#pragma unroll
    for (int b = 0; b < RADIX_BITS; ++b) {
      unsigned bal = (unsigned)__ballot((int)((d >> b) & 1u));
      mask &= ((d >> b) & 1u) ? bal : ~bal;
    }
    lr[r] = (unsigned)__popc(mask & ((1u << lane) - 1u));
    int leader = __ffs((int)mask) - 1;
    if (lane == leader) wdc[d * 128 + r * 8 + w] = (u32)__popc(mask);
  }
  __syncthreads();

  // exclusive scan of wdc[2048] in (d, r, w) order
  for (int off = 0; off < RADIX * 128; off += NTHREADS) {
    u32 v = wdc[off + tid];
    ssc[tid] = v;
    __syncthreads();
    for (int d = 1; d < NTHREADS; d <<= 1) {
      u32 t = (tid >= d) ? ssc[tid - d] : 0u;
      __syncthreads();
      ssc[tid] += t;
      __syncthreads();
    }
    u32 incl = ssc[tid];
    wdc[off + tid] = incl - v + scarry;
    __syncthreads();
    if (tid == NTHREADS - 1) scarry += incl;
    __syncthreads();
  }
  if (tid < RADIX) dstart[tid] = wdc[tid * 128];
  __syncthreads();

  for (int r = 0; r < 16; ++r) {
    unsigned rank = wdc[dg[r] * 128 + r * 8 + w] + lr[r];
    st[rank] = it[r];
  }
  __syncthreads();

  for (int k = 0; k < 16; ++k) {
    int i  = k * NTHREADS + tid;
    u64 v  = st[i];
    unsigned d = digit_of(v, shift);
    u32 pos = gOff[(size_t)d * NB + blockIdx.x] + (u32)i - dstart[d];
    outp[pos] = v;
  }
}

// ---------------------------------------------------------------------------
// Per-tile sums of fs and (vs-fs) via V_WMMA_F32_16X16X32_F16.
// A (16x32 f16): rows 0..7 (lanes with (lane&8)==0) carry fs values,
// rows 8..15 carry vs-fs. B = all-ones (layout independent). With ones B,
// D[m][j] = rowsum_m; lanes 0..15 hold rows 0..7 (fs), lanes 16..31 hold
// rows 8..15 (vs-fs); summing each lane's 8 accumulators yields the totals.
// One WMMA reduces 512 values; counts <= 512 are exact in f16/f32.
// ---------------------------------------------------------------------------
__global__ void k_partial(const u64* __restrict__ items,
                          float* __restrict__ partF, float* __restrict__ partVF) {
  __shared__ float sF[8], sV[8];
  int tid  = threadIdx.x;
  int lane = tid & 31;
  int w    = tid >> 5;
  bool fsLane  = (lane & 8) == 0;
  int  lanePos = ((lane >> 4) << 3) | (lane & 7);      // 0..15 per role
  size_t base = (size_t)blockIdx.x * TILE + (size_t)w * 512;

  v8f  acc = {};
  v16h bO;
#pragma unroll
  for (int i = 0; i < 16; ++i) bO[i] = (_Float16)1.0f;

#pragma unroll
  for (int chunk = 0; chunk < 2; ++chunk) {
    const u64* p = items + base + (size_t)chunk * 256 + (size_t)lanePos * 16;
    v16h a;
#pragma unroll
    for (int i = 0; i < 16; ++i) {
      u64 v = p[i];
      int fsb = (int)((unsigned)(v >> 1) & 1u);
      int vsb = (int)((unsigned)v & 1u);
      float val = fsLane ? (float)fsb : (float)(vsb - fsb);
      a[i] = (_Float16)val;
    }
    acc = __builtin_amdgcn_wmma_f32_16x16x32_f16(
        false, a, false, bO, (short)0, acc, false, false);
  }
  float s = acc[0] + acc[1] + acc[2] + acc[3] + acc[4] + acc[5] + acc[6] + acc[7];
  float fsT = __shfl(s, 0, 32);    // fs total (rows 0..7)
  float vfT = __shfl(s, 16, 32);   // vs-fs total (rows 8..15)
  if (lane == 0) { sF[w] = fsT; sV[w] = vfT; }
  __syncthreads();
  if (tid == 0) {
    float F = 0.f, V = 0.f;
    for (int i = 0; i < 8; ++i) { F += sF[i]; V += sV[i]; }
    partF[blockIdx.x]  = F;
    partVF[blockIdx.x] = V;
  }
}

// ---------------------------------------------------------------------------
// Exclusive scan of the NB tile partials (both arrays), emit gts (total fs).
// ---------------------------------------------------------------------------
__global__ void k_scan_partials(float* __restrict__ pF, float* __restrict__ pV,
                                float* __restrict__ gts, int c) {
  __shared__ float sA[NTHREADS], sB[NTHREADS];
  __shared__ float cA, cB;
  int tid = threadIdx.x;
  if (tid == 0) { cA = 0.f; cB = 0.f; }
  __syncthreads();
  for (int off = 0; off < NB; off += NTHREADS) {
    float a = pF[off + tid], b = pV[off + tid];
    sA[tid] = a; sB[tid] = b;
    __syncthreads();
    for (int d = 1; d < NTHREADS; d <<= 1) {
      float ta = (tid >= d) ? sA[tid - d] : 0.f;
      float tb = (tid >= d) ? sB[tid - d] : 0.f;
      __syncthreads();
      sA[tid] += ta; sB[tid] += tb;
      __syncthreads();
    }
    float iA = sA[tid], iB = sB[tid];
    pF[off + tid] = iA - a + cA;
    pV[off + tid] = iB - b + cB;
    __syncthreads();
    if (tid == NTHREADS - 1) { cA += iA; cB += iB; }
    __syncthreads();
  }
  if (tid == 0) gts[c] = cA;
}

__device__ __forceinline__ float jaccf(float G, float F, float V) {
  float inter = G - F;
  float uni   = G + V;
  float den   = (uni == 0.f) ? 1.f : uni;
  return 1.f - inter / den;
}

// ---------------------------------------------------------------------------
// loss_c = sum_i es_i * (jacc_i - jacc_{i-1}) over the sorted sequence.
// Cumulative counts = global tile base (scanned partials) + in-block scan.
// Deterministic: per-block sums written to lossBlk, reduced in fixed order.
// ---------------------------------------------------------------------------
__global__ void k_loss(const u64* __restrict__ items,
                       const float* __restrict__ partF,
                       const float* __restrict__ partVF,
                       const float* __restrict__ gts,
                       float* __restrict__ lossBlk, int c) {
  __shared__ float sF[NTHREADS], sV[NTHREADS];
  int tid = threadIdx.x;
  float G = gts[c];
  size_t base = (size_t)blockIdx.x * TILE + (size_t)tid * 16;

  float fsum = 0.f, vsum = 0.f;
#pragma unroll
  for (int k = 0; k < 16; ++k) {
    u64 v = items[base + k];
    int fsb = (int)((unsigned)(v >> 1) & 1u);
    int vsb = (int)((unsigned)v & 1u);
    fsum += (float)fsb;
    vsum += (float)(vsb - fsb);
  }
  sF[tid] = fsum; sV[tid] = vsum;
  __syncthreads();
  for (int d = 1; d < NTHREADS; d <<= 1) {
    float tf = (tid >= d) ? sF[tid - d] : 0.f;
    float tv = (tid >= d) ? sV[tid - d] : 0.f;
    __syncthreads();
    sF[tid] += tf; sV[tid] += tv;
    __syncthreads();
  }
  float F = partF[blockIdx.x]  + sF[tid] - fsum;   // exclusive cum fs
  float V = partVF[blockIdx.x] + sV[tid] - vsum;   // exclusive cum (vs-fs)

  float acc = 0.f;
#pragma unroll
  for (int k = 0; k < 16; ++k) {
    u64 v   = items[base + k];
    int fsb = (int)((unsigned)(v >> 1) & 1u);
    int vsb = (int)((unsigned)v & 1u);
    float es = __uint_as_float(~(u32)(v >> 2));
    float Fi = F + (float)fsb;
    float Vi = V + (float)(vsb - fsb);
    acc += es * (jaccf(G, Fi, Vi) - jaccf(G, F, V));
    F = Fi; V = Vi;
  }
  __syncthreads();
  sF[tid] = acc;
  __syncthreads();
  for (int s2 = NTHREADS / 2; s2 > 0; s2 >>= 1) {
    if (tid < s2) sF[tid] += sF[tid + s2];
    __syncthreads();
  }
  if (tid == 0) lossBlk[blockIdx.x] = sF[0];
}

__global__ void k_reduce_loss(const float* __restrict__ lossBlk,
                              float* __restrict__ lossAcc, int c) {
  __shared__ float s[NTHREADS];
  int tid = threadIdx.x;
  float a = 0.f;
  for (int i = tid; i < NB; i += NTHREADS) a += lossBlk[i];
  s[tid] = a;
  __syncthreads();
  for (int s2 = NTHREADS / 2; s2 > 0; s2 >>= 1) {
    if (tid < s2) s[tid] += s[tid + s2];
    __syncthreads();
  }
  if (tid == 0) lossAcc[c] = s[0];
}

__global__ void k_finalize(const float* __restrict__ lossAcc,
                           const float* __restrict__ gts,
                           float* __restrict__ out) {
  int t = threadIdx.x;
  float l = 0.f, p = 0.f;
  if (t < NUM_CLASSES) {
    p = (gts[t] > 0.f) ? 1.f : 0.f;
    l = lossAcc[t] * p;
  }
  for (int off = 16; off > 0; off >>= 1) {
    l += __shfl_down(l, off, 32);
    p += __shfl_down(p, off, 32);
  }
  if (t == 0) out[0] = l / fmaxf(p, 1.f);
}

// ---------------------------------------------------------------------------
extern "C" void kernel_launch(void* const* d_in, const int* in_sizes, int n_in,
                              void* d_out, int out_size, void* d_ws, size_t ws_size,
                              hipStream_t stream) {
  const float*     logits = (const float*)d_in[0];
  const long long* labels = (const long long*)d_in[1];
  float* out = (float*)d_out;
  char*  ws  = (char*)d_ws;

  u64* items0  = (u64*)(ws + OFF_ITEMS0);
  u64* items1  = (u64*)(ws + OFF_ITEMS1);
  u32* gOff    = (u32*)(ws + OFF_GOFF);
  float* mmax  = (float*)(ws + OFF_MMAX);
  float* msum  = (float*)(ws + OFF_MSUM);
  unsigned char* lab8 = (unsigned char*)(ws + OFF_LAB8);
  float* partF   = (float*)(ws + OFF_PARTF);
  float* partVF  = (float*)(ws + OFF_PARTVF);
  float* lossBlk = (float*)(ws + OFF_LOSSBLK);
  float* gts     = (float*)(ws + OFF_GTS);
  float* lossAcc = (float*)(ws + OFF_LOSSACC);

  k_prep<<<NPIX / NTHREADS, NTHREADS, 0, stream>>>(logits, labels, mmax, msum, lab8);

  for (int c = 0; c < NUM_CLASSES; ++c) {
    k_build<<<NPIX / NTHREADS, NTHREADS, 0, stream>>>(logits, mmax, msum, lab8,
                                                      items0, c);
    u64* src = items0;
    u64* dst = items1;
    for (int p = 0; p < N_PASSES; ++p) {
      int shift = 2 + RADIX_BITS * p;
      k_hist<<<NB, NTHREADS, 0, stream>>>(src, gOff, shift);
      k_scan_u32<<<1, NTHREADS, 0, stream>>>(gOff, RADIX * NB);
      k_scatter<<<NB, NTHREADS, 0, stream>>>(src, dst, gOff, shift);
      u64* t = src; src = dst; dst = t;
    }
    // 8 passes -> sorted data back in items0 (== src)
    k_partial<<<NB, NTHREADS, 0, stream>>>(src, partF, partVF);
    k_scan_partials<<<1, NTHREADS, 0, stream>>>(partF, partVF, gts, c);
    k_loss<<<NB, NTHREADS, 0, stream>>>(src, partF, partVF, gts, lossBlk, c);
    k_reduce_loss<<<1, NTHREADS, 0, stream>>>(lossBlk, lossAcc, c);
  }
  k_finalize<<<1, 32, 0, stream>>>(lossAcc, gts, out);
}
